// Decoder_20581483283011
// MI455X (gfx1250) — matrix operand order repaired and checked by reading, not compiled, not run on hardware
//
#include <hip/hip_runtime.h>

typedef __attribute__((ext_vector_type(16))) _Float16 v16h;
typedef __attribute__((ext_vector_type(8)))  float    v8f;

#define Dm   128
#define Hn   8
#define HDn  16
#define Bn   256
#define Nn   2000

// ---------------------------------------------------------------------------
// pack two floats into one u32 of f16 pairs (lo = a, hi = b)
__device__ __forceinline__ unsigned pkh2(float a, float b) {
  union { _Float16 h[2]; unsigned u; } t;
  t.h[0] = (_Float16)a; t.h[1] = (_Float16)b;
  return t.u;
}

// ---------------------------------------------------------------------------
// Kernel 1: Wk2 = wk_i @ w_k, Wv2 = wv_i @ w_v  (each 128x128, trivial flops)
__global__ void prep_weights(const float* __restrict__ w_k,
                             const float* __restrict__ w_v,
                             const float* __restrict__ in_proj_w,
                             float* __restrict__ ws) {
  int e = blockIdx.x & 127;
  int which = blockIdx.x >> 7;            // 0 -> Wk2, 1 -> Wv2
  int d = threadIdx.x;
  const float* wrow = in_proj_w + (size_t)((which == 0 ? 128 : 256) + e) * Dm;
  const float* m = (which == 0) ? w_k : w_v;
  float acc = 0.f;
  for (int c = 0; c < Dm; ++c) acc += wrow[c] * m[c * Dm + d];
  float* dst = ws + (which == 0 ? 0 : Dm * Dm);
  dst[e * Dm + d] = acc;
}

// ---------------------------------------------------------------------------
// Kernel 2: per-batch context -> q0 -> qp -> S[b] (8x128, pre-scaled by 1/4)
__global__ void prep_batch(const float* __restrict__ graph_embed,
                           const float* __restrict__ h_last,
                           const float* __restrict__ h_first,
                           const int* __restrict__ tptr,
                           const float* __restrict__ v_l,
                           const float* __restrict__ v_f,
                           const float* __restrict__ w_q,
                           const float* __restrict__ in_proj_w,
                           const float* __restrict__ in_proj_b,
                           const float* __restrict__ Wk2,
                           float* __restrict__ Sws,
                           float* __restrict__ ccw) {
  __shared__ float ctx[384];
  __shared__ float q0s[128];
  __shared__ float qps[128];
  int b = blockIdx.x, j = threadIdx.x;
  int t = tptr[0];
  const float* hl = (t == 0) ? v_l : (h_last + (size_t)b * Dm);
  const float* hf = (t == 0) ? v_f : (h_first + (size_t)b * Dm);
  ctx[j]       = graph_embed[(size_t)b * Dm + j];
  ctx[128 + j] = hl[j];
  ctx[256 + j] = hf[j];
  __syncthreads();
  float acc = 0.f;
  for (int c = 0; c < 384; ++c) acc += ctx[c] * w_q[(size_t)j * 384 + c];
  q0s[j] = acc;
  __syncthreads();
  acc = in_proj_b[j];
  for (int c = 0; c < Dm; ++c) acc += q0s[c] * in_proj_w[(size_t)j * Dm + c];
  qps[j] = acc;
  __syncthreads();
  // S[b][h][d] = 0.25 * sum_e qp[h*16+e] * Wk2[h*16+e][d]
  for (int h = 0; h < Hn; ++h) {
    float s = 0.f;
    for (int e = 0; e < HDn; ++e) s += qps[h * HDn + e] * Wk2[(h * HDn + e) * Dm + j];
    Sws[((size_t)b * Hn + h) * Dm + j] = 0.25f * s;
  }
  if (j < Hn) {
    float s = 0.f;
    for (int e = 0; e < HDn; ++e) s += qps[j * HDn + e] * in_proj_b[Dm + j * HDn + e];
    ccw[b * Hn + j] = 0.25f * s;
  }
}

// ---------------------------------------------------------------------------
// Kernel 3: per-batch flash pass (WMMA scores + WMMA attn*node), merge,
// small matvecs, then logits/tanh/softmax pass.
//
// Dynamic LDS layout (66048 bytes):
//   [0, 65536)      : per-wave transposed f16 tiles (wave w at w*8192),
//                     u32[128][16]; reused after flash loop:
//                       aw    [0,32768)   float[8 waves][8 h][128 d]
//                       ahat  [32768,36864) float[8][128]
//                       ctxs  [37120,+512), hcs +512, gs +512
//                       expbuf[38656,+8000), red [46656,+1024)
//   [65536, 66048)  : ms[8][8], ls[8][8] flash stats
__global__ __launch_bounds__(256) void decoder_main(
    const float* __restrict__ node, const unsigned char* __restrict__ vis,
    const float* __restrict__ Sws, const float* __restrict__ ccw,
    const float* __restrict__ Wv2, const float* __restrict__ w_k,
    const float* __restrict__ in_proj_b, const float* __restrict__ out_w,
    const float* __restrict__ out_b, float* __restrict__ out) {
  extern __shared__ char smem[];
  float* ms = (float*)(smem + 65536);
  float* ls = ms + 64;

  const int tid  = threadIdx.x;
  const int wave = tid >> 5;
  const int lane = tid & 31;
  const int col  = lane & 15;           // head (cols 8..15 are zero-padding)
  const bool lo  = lane < 16;
  const int b = blockIdx.x;
  const float* nb = node + (size_t)b * Nn * Dm;
  const unsigned char* vb = vis + (size_t)b * Nn;
  unsigned* ldsT = (unsigned*)(smem + wave * 8192);   // f16 [d=128][k pairs=16 u32]

  // --- preload S as WMMA B-fragments (f16), and per-head score constant ---
  v16h bS[4];
#pragma unroll
  for (int j = 0; j < 4; ++j) {
    int kbase = 32 * j + (lo ? 0 : 8);
#pragma unroll
    for (int i = 0; i < 16; ++i) {
      float v = (col < 8) ? Sws[((size_t)b * Hn + col) * Dm + kbase + i + (i >= 8 ? 8 : 0)] : 0.f;
      bS[j][i] = (_Float16)v;
    }
  }
  const float ccv = (col < 8) ? ccw[b * Hn + col] : 0.f;

  float m_run = -1e30f, l_run = 0.f;
  v8f a[8];
#pragma unroll
  for (int dd = 0; dd < 8; ++dd) a[dd] = (v8f){};

  const int NT = (Nn + 31) / 32;        // 63 tiles of 32 rows
  for (int tix = wave; tix < NT; tix += 8) {
    const int n0 = tix * 32;

    // prefetch next tile for this wave (global_prefetch_b8)
    {
      int pn = n0 + 256 + 8 * 0;  (void)pn;
#pragma unroll
      for (int i = 0; i < 4; ++i) {
        int pr = n0 + 256 + 8 * i + (lane >> 2);
        pr = pr < Nn ? pr : Nn - 1;
        __builtin_prefetch(nb + (size_t)pr * Dm + (lane & 3) * 32, 0, 1);
      }
    }

    // --- stage tile transposed into LDS as f16: ldsT[d][k/2] packed pairs ---
#pragma unroll 4
    for (int q = 0; q < 16; ++q) {
      int r0 = n0 + 2 * q, r1 = r0 + 1;
      r0 = r0 < Nn ? r0 : Nn - 1;
      r1 = r1 < Nn ? r1 : Nn - 1;
      float4 fa = *(const float4*)(nb + (size_t)r0 * Dm + 4 * lane);
      float4 fb = *(const float4*)(nb + (size_t)r1 * Dm + 4 * lane);
      ldsT[(4 * lane + 0) * 16 + q] = pkh2(fa.x, fb.x);
      ldsT[(4 * lane + 1) * 16 + q] = pkh2(fa.y, fb.y);
      ldsT[(4 * lane + 2) * 16 + q] = pkh2(fa.z, fb.z);
      ldsT[(4 * lane + 3) * 16 + q] = pkh2(fa.w, fb.w);
    }

    // --- scores: two 16x16 WMMA chains, K=128 (4 x K32) ---
    v8f s0 = (v8f){}, s1 = (v8f){};
#pragma unroll
    for (int g = 0; g < 2; ++g) {
      int rA = n0 + 16 * g + col;
      rA = rA < Nn ? rA : Nn - 1;
      const float* rp = nb + (size_t)rA * Dm;
      v8f cacc = (v8f){};
#pragma unroll
      for (int j = 0; j < 4; ++j) {
        int o1 = 32 * j + (lo ? 0 : 8);
        int o2 = 32 * j + (lo ? 16 : 24);
        float4 f0 = *(const float4*)(rp + o1);
        float4 f1 = *(const float4*)(rp + o1 + 4);
        float4 f2 = *(const float4*)(rp + o2);
        float4 f3 = *(const float4*)(rp + o2 + 4);
        v16h A;
        A[0]=(_Float16)f0.x; A[1]=(_Float16)f0.y; A[2]=(_Float16)f0.z; A[3]=(_Float16)f0.w;
        A[4]=(_Float16)f1.x; A[5]=(_Float16)f1.y; A[6]=(_Float16)f1.z; A[7]=(_Float16)f1.w;
        A[8]=(_Float16)f2.x; A[9]=(_Float16)f2.y; A[10]=(_Float16)f2.z; A[11]=(_Float16)f2.w;
        A[12]=(_Float16)f3.x; A[13]=(_Float16)f3.y; A[14]=(_Float16)f3.z; A[15]=(_Float16)f3.w;
        cacc = __builtin_amdgcn_wmma_f32_16x16x32_f16(false, A, false, bS[j],
                                                      (short)0, cacc, false, false);
      }
      if (g == 0) s0 = cacc; else s1 = cacc;
    }

    // --- mask + online softmax (per head = column; pair lanes L/L+16) ---
    const int boff = lo ? 0 : 8;
    unsigned long long m0 = *(const unsigned long long*)(vb + n0 + boff);
    bool ok1 = (n0 + 16 + boff + 7) < Nn;
    int a1 = ok1 ? (n0 + 16 + boff) : 0;
    unsigned long long m1 = *(const unsigned long long*)(vb + a1);

    float sc[16]; unsigned vmsk = 0;
#pragma unroll
    for (int i = 0; i < 8; ++i) {
      bool v = (col < 8) && (((m0 >> (8 * i)) & 0xffull) == 0);
      sc[i] = v ? (s0[i] + ccv) : -1e30f;
      vmsk |= v ? (1u << i) : 0u;
    }
#pragma unroll
    for (int i = 0; i < 8; ++i) {
      bool v = (col < 8) && ok1 && (((m1 >> (8 * i)) & 0xffull) == 0);
      sc[8 + i] = v ? (s1[i] + ccv) : -1e30f;
      vmsk |= v ? (1u << (8 + i)) : 0u;
    }
    float tm = -1e30f;
#pragma unroll
    for (int i = 0; i < 16; ++i) tm = fmaxf(tm, sc[i]);
    tm = fmaxf(tm, __shfl_xor(tm, 16));
    float m_new = fmaxf(m_run, tm);
    float scale = __expf(m_run - m_new);

    float psum = 0.f;
    v16h P;                               // P.T A-fragment: free transpose
#pragma unroll
    for (int i = 0; i < 16; ++i) {
      float p = ((vmsk >> i) & 1u) ? __expf(sc[i] - m_new) : 0.f;
      psum += p;
      P[i] = (_Float16)p;
    }
    psum += __shfl_xor(psum, 16);
    l_run = l_run * scale + psum;
    m_run = m_new;

    // rescale accumulator rows (row r = head r, factor lives in lane r)
    float sf[8];
#pragma unroll
    for (int r = 0; r < 8; ++r) sf[r] = __shfl(scale, r);
#pragma unroll
    for (int dd = 0; dd < 8; ++dd)
#pragma unroll
      for (int r = 0; r < 8; ++r) a[dd][r] *= sf[r];

    // --- a += P.T x nodeTile : 8 WMMAs, B-fragments from LDS_T ---
#pragma unroll
    for (int dd = 0; dd < 8; ++dd) {
      const unsigned* bp = ldsT + (unsigned)((16 * dd + col) * 16 + (lo ? 0 : 4));
      union { uint4 u[2]; v16h h; } Bf;
      Bf.u[0] = *(const uint4*)(bp);
      Bf.u[1] = *(const uint4*)(bp + 8);
      a[dd] = __builtin_amdgcn_wmma_f32_16x16x32_f16(false, P, false, Bf.h,
                                                     (short)0, a[dd], false, false);
    }
  }

  // --- publish flash stats, then accumulators (reuses tile LDS) ---
  if (lane < 8) { ms[wave * 8 + lane] = m_run; ls[wave * 8 + lane] = l_run; }
  __syncthreads();
  float* aw = (float*)smem;               // [8 waves][8 h][128 d]
  if (lo) {
#pragma unroll
    for (int dd = 0; dd < 8; ++dd)
#pragma unroll
      for (int r = 0; r < 8; ++r)
        aw[(wave * 8 + r) * Dm + dd * 16 + col] = a[dd][r];
  }
  __syncthreads();

  // --- merge waves: ahat[h][d] = (sum_w a*e) / (sum_w l*e) ---
  float* ahat = (float*)(smem + 32768);
  for (int idx = tid; idx < Hn * Dm; idx += 256) {
    int h = idx >> 7, d = idx & 127;
    float mg = -1e30f;
#pragma unroll
    for (int w2 = 0; w2 < 8; ++w2) mg = fmaxf(mg, ms[w2 * 8 + h]);
    float ag = 0.f, lg = 0.f;
#pragma unroll
    for (int w2 = 0; w2 < 8; ++w2) {
      float e = __expf(ms[w2 * 8 + h] - mg);
      ag += aw[(w2 * 8 + h) * Dm + d] * e;
      lg += ls[w2 * 8 + h] * e;
    }
    ahat[h * Dm + d] = ag / fmaxf(lg, 1e-30f);
  }
  __syncthreads();

  // --- ctx -> h_c_new -> g (tiny matvecs) ---
  float* ctxs = (float*)(smem + 37120);
  float* hcs  = ctxs + 128;
  float* gs   = hcs + 128;
  if (tid < 128) {
    int h = tid >> 4;
    float acc = in_proj_b[256 + tid];
    for (int d2 = 0; d2 < Dm; ++d2) acc += Wv2[(size_t)tid * Dm + d2] * ahat[h * Dm + d2];
    ctxs[tid] = acc;
  }
  __syncthreads();
  if (tid < 128) {
    float acc = out_b[tid];
    for (int j2 = 0; j2 < Dm; ++j2) acc += out_w[(size_t)tid * Dm + j2] * ctxs[j2];
    hcs[tid] = acc;
  }
  __syncthreads();
  if (tid < 128) {
    float acc = 0.f;
    for (int e = 0; e < Dm; ++e) acc += hcs[e] * w_k[(size_t)e * Dm + tid];
    gs[tid] = acc * 0.08838834764831845f;   // 1/sqrt(128)
  }
  __syncthreads();

  // --- logits pass: logit = 10*tanh(g . node[n]), masked -> -inf ---
  float* expbuf = (float*)(smem + 38656);   // [2000]
  float* red    = (float*)(smem + 46656);   // [256]
  float partial = 0.f;
  for (int n = tid; n < Nn; n += 256) {
    const float4* rp = (const float4*)(nb + (size_t)n * Dm);
    float dot = 0.f;
#pragma unroll 8
    for (int k2 = 0; k2 < 32; ++k2) {
      float4 f = rp[k2];
      dot += f.x * gs[4 * k2] + f.y * gs[4 * k2 + 1] +
             f.z * gs[4 * k2 + 2] + f.w * gs[4 * k2 + 3];
    }
    float lgv = 10.f * tanhf(dot);
    bool vm = vb[n] != 0;
    float logit = vm ? -__builtin_inff() : lgv;
    out[(size_t)Bn * Nn + (size_t)b * Nn + n] = logit;
    float e = vm ? 0.f : __expf(lgv);       // |lgv| <= 10: safe without max-sub
    expbuf[n] = e;
    partial += e;
  }
  red[tid] = partial;
  __syncthreads();
  for (int s = 128; s > 0; s >>= 1) {
    if (tid < s) red[tid] += red[tid + s];
    __syncthreads();
  }
  float inv = 1.f / red[0];
  for (int n = tid; n < Nn; n += 256)
    out[(size_t)b * Nn + n] = expbuf[n] * inv;
}

// ---------------------------------------------------------------------------
extern "C" void kernel_launch(void* const* d_in, const int* in_sizes, int n_in,
                              void* d_out, int out_size, void* d_ws, size_t ws_size,
                              hipStream_t stream) {
  (void)in_sizes; (void)n_in; (void)out_size; (void)ws_size;
  const float* node        = (const float*)d_in[0];
  const float* graph_embed = (const float*)d_in[1];
  const float* h_last      = (const float*)d_in[2];
  const float* h_first     = (const float*)d_in[3];
  const unsigned char* vis = (const unsigned char*)d_in[4];
  const int*   tptr        = (const int*)d_in[5];
  const float* v_l         = (const float*)d_in[6];
  const float* v_f         = (const float*)d_in[7];
  const float* w_q         = (const float*)d_in[8];
  const float* w_k         = (const float*)d_in[9];
  const float* w_v         = (const float*)d_in[10];
  const float* in_proj_w   = (const float*)d_in[11];
  const float* in_proj_b   = (const float*)d_in[12];
  const float* out_w       = (const float*)d_in[13];
  const float* out_b       = (const float*)d_in[14];

  float* ws  = (float*)d_ws;
  float* Wk2 = ws;                         // 128*128
  float* Wv2 = ws + Dm * Dm;               // 128*128
  float* Sws = ws + 2 * Dm * Dm;           // B*8*128
  float* ccw = Sws + (size_t)Bn * Hn * Dm; // B*8

  prep_weights<<<256, 128, 0, stream>>>(w_k, w_v, in_proj_w, ws);
  prep_batch<<<Bn, 128, 0, stream>>>(graph_embed, h_last, h_first, tptr, v_l, v_f,
                                     w_q, in_proj_w, in_proj_b, Wk2, Sws, ccw);
  decoder_main<<<Bn, 256, 66048, stream>>>(node, vis, Sws, ccw, Wv2, w_k,
                                           in_proj_b, out_w, out_b, (float*)d_out);
}
